// CausalSelfAttention_13408887898884
// MI455X (gfx1250) — matrix operand order repaired and checked by reading, not compiled, not run on hardware
//
#include <hip/hip_runtime.h>
#include <stdint.h>

typedef __bf16 bf16_t;
typedef __attribute__((ext_vector_type(16))) __bf16 v16bf;
typedef __attribute__((ext_vector_type(8)))  float   v8f;
typedef int v4i_t __attribute__((__vector_size__(16)));

#define BATCH   4
#define NSEQ    2048
#define DMODEL  1024
#define NHEAD   16
#define HDIM    64
#define MROWS   (BATCH * NSEQ)   // 8192
#define N3      (3 * DMODEL)     // 3072

#if __has_builtin(__builtin_amdgcn_global_load_async_to_lds_b128)
#define HAVE_ASYNC_LDS 1
#endif

union FragU {
    v16bf v;
    uint4 q[2];
};

static __device__ __forceinline__ v8f v8f_zero() {
    v8f z;
#pragma unroll
    for (int e = 0; e < 8; ++e) z[e] = 0.0f;
    return z;
}

static __device__ __forceinline__ v8f wmma_bf16(const FragU& a, const FragU& b, v8f c) {
    return __builtin_amdgcn_wmma_f32_16x16x32_bf16(false, a.v, false, b.v,
                                                   (short)0, c, false, false);
}

// 16B global -> LDS copy: async DMA path on CDNA5 (ASYNCcnt), sync fallback.
static __device__ __forceinline__ void cp16_g2l(void* lds, const void* g) {
#ifdef HAVE_ASYNC_LDS
    __builtin_amdgcn_global_load_async_to_lds_b128(
        (__attribute__((address_space(1))) v4i_t*)g,
        (__attribute__((address_space(3))) v4i_t*)lds, 0, 0);
#else
    *(uint4*)lds = *(const uint4*)g;
#endif
}

// wait until all async copies issued by this wave have landed
static __device__ __forceinline__ void wait_async0() {
#ifdef HAVE_ASYNC_LDS
#if __has_builtin(__builtin_amdgcn_s_wait_asynccnt)
    __builtin_amdgcn_s_wait_asynccnt(0);
#else
    asm volatile("s_wait_asynccnt 0" ::: "memory");
#endif
#endif
}

// wait until <= 8 async copies outstanding (previous double-buffer stage done,
// current stage's 8 copies may still be in flight)
static __device__ __forceinline__ void wait_async8() {
#ifdef HAVE_ASYNC_LDS
#if __has_builtin(__builtin_amdgcn_s_wait_asynccnt)
    __builtin_amdgcn_s_wait_asynccnt(8);
#else
    asm volatile("s_wait_asynccnt 8" ::: "memory");
#endif
#endif
}

// ---------------------------------------------------------------------------
// fp32 -> bf16 elementwise
// ---------------------------------------------------------------------------
__global__ void k_f32_to_bf16(const float* __restrict__ in, bf16_t* __restrict__ out, int n) {
    int i = blockIdx.x * blockDim.x + threadIdx.x;
    if (i < n) out[i] = (bf16_t)in[i];
}

// in: [rows][cols] fp32  ->  out: [cols][rows] bf16 (transpose + convert)
__global__ void k_transpose_bf16(const float* __restrict__ in, bf16_t* __restrict__ out,
                                 int rows, int cols) {
    int i = blockIdx.x * blockDim.x + threadIdx.x;
    if (i >= rows * cols) return;
    int orow = i / rows;   // = original col
    int ocol = i % rows;   // = original row
    out[(size_t)orow * rows + ocol] = (bf16_t)in[(size_t)ocol * cols + orow];
}

// ---------------------------------------------------------------------------
// Tiled bf16 WMMA GEMM:  C[M][N] = A[M][K] * Bt[N][K]^T + bias[N]
// Block tile 128x128, BK=64 (2 WMMA K-chunks per stage), double-buffered LDS
// with async global->LDS copies. 256 threads = 8 waves, each wave 32(m)x64(n).
// MODE 0: scatter to Q [B,H,N,HD], K [B,H,N,HD], V^T [B,H,HD,N] (bf16)
// MODE 1: store fp32 to Cf [M][N]
// ---------------------------------------------------------------------------
template <int MODE>
__global__ __launch_bounds__(256)
void k_gemm(const bf16_t* __restrict__ A, const bf16_t* __restrict__ Bt,
            const float* __restrict__ bias, float* __restrict__ Cf,
            bf16_t* __restrict__ Qb, bf16_t* __restrict__ Kb, bf16_t* __restrict__ Vtb,
            int M, int N, int K) {
    __shared__ bf16_t Asm[2][128][72];   // +8 pad (144B stride) breaks bank conflicts
    __shared__ bf16_t Bsm[2][128][72];

    const int tid  = threadIdx.x;
    const int wv   = tid >> 5;
    const int lane = tid & 31;
    const int half = lane >> 4;
    const int lo   = lane & 15;
    const int wm   = wv & 3;          // 4 waves along M
    const int wn   = wv >> 2;         // 2 waves along N
    const int n0   = blockIdx.x * 128;
    const int m0   = blockIdx.y * 128;

    v8f acc[2][4];
#pragma unroll
    for (int mi = 0; mi < 2; ++mi)
#pragma unroll
        for (int ni = 0; ni < 4; ++ni) acc[mi][ni] = v8f_zero();

    const int lrow = tid >> 1;          // 0..127
    const int lseg = (tid & 1) << 5;    // 0 or 32 elements (64B)

    // issue the 8 per-thread 16B copies for one 128x64 A/B stage
    auto issue_tile = [&](int k0, int bi) {
        const bf16_t* ga = A + (size_t)(m0 + lrow) * K + k0 + lseg;
        const bf16_t* gb = Bt + (size_t)(n0 + lrow) * K + k0 + lseg;
#pragma unroll
        for (int u = 0; u < 4; ++u)
            cp16_g2l(&Asm[bi][lrow][lseg + u * 8], ga + u * 8);
#pragma unroll
        for (int u = 0; u < 4; ++u)
            cp16_g2l(&Bsm[bi][lrow][lseg + u * 8], gb + u * 8);
    };

    issue_tile(0, 0);
    const int KT = K >> 6;
    for (int kt = 0; kt < KT; ++kt) {
        const int bi = kt & 1;
        if (kt + 1 < KT) {
            issue_tile((kt + 1) << 6, bi ^ 1);  // prefetch overlaps WMMAs below
            wait_async8();                      // previous stage has landed
        } else {
            wait_async0();
        }
        __syncthreads();

#pragma unroll
        for (int kk = 0; kk < 2; ++kk) {
            FragU af[2], bf[4];
#pragma unroll
            for (int mi = 0; mi < 2; ++mi) {
                // A-frag: lane = row, K split {half*8.., 16+half*8..}
                const bf16_t* p = &Asm[bi][wm * 32 + mi * 16 + lo][kk * 32];
                af[mi].q[0] = *(const uint4*)(p + half * 8);
                af[mi].q[1] = *(const uint4*)(p + 16 + half * 8);
            }
#pragma unroll
            for (int ni = 0; ni < 4; ++ni) {
                // B-frag: lane = out column, K split {half*8.., 16+half*8..}
                const bf16_t* p = &Bsm[bi][wn * 64 + ni * 16 + lo][kk * 32 + half * 8];
                bf[ni].q[0] = *(const uint4*)(p);
                bf[ni].q[1] = *(const uint4*)(p + 16);
            }
#pragma unroll
            for (int mi = 0; mi < 2; ++mi)
#pragma unroll
                for (int ni = 0; ni < 4; ++ni)
                    acc[mi][ni] = wmma_bf16(af[mi], bf[ni], acc[mi][ni]);
        }
        __syncthreads();   // all waves done reading buf bi before it is refilled
    }

    // Epilogue: D layout -> element (M = r + 8*half, N = lo)
#pragma unroll
    for (int mi = 0; mi < 2; ++mi) {
#pragma unroll
        for (int ni = 0; ni < 4; ++ni) {
#pragma unroll
            for (int r = 0; r < 8; ++r) {
                const int m = m0 + wm * 32 + mi * 16 + r + half * 8;
                const int c = n0 + wn * 64 + ni * 16 + lo;
                const float v = acc[mi][ni][r] + bias[c];
                if (MODE == 1) {
                    Cf[(size_t)m * N + c] = v;
                } else {
                    const bf16_t bv = (bf16_t)v;
                    const int b = m >> 11, n = m & (NSEQ - 1);
                    const int which = c >> 10, cc = c & (DMODEL - 1);
                    const int h = cc >> 6, hd = cc & (HDIM - 1);
                    if (which == 0)
                        Qb[(((size_t)b * NHEAD + h) * NSEQ + n) * HDIM + hd] = bv;
                    else if (which == 1)
                        Kb[(((size_t)b * NHEAD + h) * NSEQ + n) * HDIM + hd] = bv;
                    else
                        Vtb[(((size_t)b * NHEAD + h) * HDIM + hd) * NSEQ + n] = bv;
                }
            }
        }
    }
}

// ---------------------------------------------------------------------------
// Flash attention (causal, bf16 WMMA, fp32 online softmax)
// grid = (NSEQ/64, B*H), block = 128 (4 waves). Wave handles 16 query rows.
// K tile [64 keys][64 hd], V^T tile [64 hd][64 keys]: double-buffered in LDS,
// filled with async global->LDS copies that overlap the previous block's math.
// ---------------------------------------------------------------------------
__global__ __launch_bounds__(128)
void k_flash(const bf16_t* __restrict__ Qb, const bf16_t* __restrict__ Kb,
             const bf16_t* __restrict__ Vtb, bf16_t* __restrict__ AOb) {
    __shared__ bf16_t Ksm[2][64][72];    // [key][hd], +8 pad
    __shared__ bf16_t Vsm[2][64][72];    // [hd][key], +8 pad
    __shared__ bf16_t Psm[4][16][72];    // per-wave P staging (16 q rows x 64 keys)

    const int tid  = threadIdx.x;
    const int wv   = tid >> 5;
    const int lane = tid & 31;
    const int half = lane >> 4;
    const int lo   = lane & 15;
    const int qx   = blockIdx.x;          // query block (64 rows)
    const int bh   = blockIdx.y;          // b*H + h
    const int b    = bh >> 4;
    const int h    = bh & (NHEAD - 1);
    const int q0   = qx * 64 + wv * 16;   // first query row of this wave

    const bf16_t* Qh = Qb  + (size_t)bh * NSEQ * HDIM;
    const bf16_t* Kh = Kb  + (size_t)bh * NSEQ * HDIM;
    const bf16_t* Vh = Vtb + (size_t)bh * HDIM * NSEQ;

    // Q A-fragments, kept in registers for the whole key loop (HD=64 -> 2 chunks)
    FragU aq[2];
    {
        const bf16_t* qrow = Qh + (size_t)(q0 + lo) * HDIM;
#pragma unroll
        for (int kk = 0; kk < 2; ++kk) {
            aq[kk].q[0] = *(const uint4*)(qrow + kk * 32 + half * 8);
            aq[kk].q[1] = *(const uint4*)(qrow + kk * 32 + 16 + half * 8);
        }
    }

    v8f accO[4];
#pragma unroll
    for (int n = 0; n < 4; ++n) accO[n] = v8f_zero();
    float mrun[8], lrun[8];
#pragma unroll
    for (int r = 0; r < 8; ++r) { mrun[r] = -3.0e38f; lrun[r] = 0.0f; }

    const float sc = 0.125f;   // HD^-0.5

    const int lrow = tid >> 1;         // 0..63
    const int lseg = (tid & 1) << 5;   // 0 or 32 elements (64B)

    // 8 per-thread 16B copies per K/V stage
    auto issue_kv = [&](int kb, int bi) {
        const bf16_t* gk = Kh + (size_t)(kb * 64 + lrow) * HDIM + lseg;
        const bf16_t* gv = Vh + (size_t)lrow * NSEQ + kb * 64 + lseg;
#pragma unroll
        for (int u = 0; u < 4; ++u)
            cp16_g2l(&Ksm[bi][lrow][lseg + u * 8], gk + u * 8);
#pragma unroll
        for (int u = 0; u < 4; ++u)
            cp16_g2l(&Vsm[bi][lrow][lseg + u * 8], gv + u * 8);
    };

    issue_kv(0, 0);
    for (int kb = 0; kb <= qx; ++kb) {   // causal: skip fully-masked key blocks
        const int bi = kb & 1;
        if (kb < qx) {
            issue_kv(kb + 1, bi ^ 1);    // prefetch next K/V block
            wait_async8();
        } else {
            wait_async0();
        }
        __syncthreads();

        // S = Q * K^T  (4 key sub-tiles x 2 K-chunks)
        v8f s[4];
#pragma unroll
        for (int j = 0; j < 4; ++j) s[j] = v8f_zero();
#pragma unroll
        for (int j = 0; j < 4; ++j) {
#pragma unroll
            for (int kk = 0; kk < 2; ++kk) {
                FragU bk;
                const bf16_t* p = &Ksm[bi][j * 16 + lo][kk * 32 + half * 8];
                bk.q[0] = *(const uint4*)(p);
                bk.q[1] = *(const uint4*)(p + 16);
                s[j] = wmma_bf16(aq[kk], bk, s[j]);
            }
        }

        // scale (+ causal mask only on the diagonal block)
        if (kb == qx) {
            const int kbase = kb * 64;
#pragma unroll
            for (int j = 0; j < 4; ++j) {
                const int key = kbase + j * 16 + lo;
#pragma unroll
                for (int r = 0; r < 8; ++r) {
                    const int q = q0 + r + half * 8;
                    float v = s[j][r] * sc;
                    if (key > q) v -= 1.0e9f;   // same additive -1e9 as reference
                    s[j][r] = v;
                }
            }
        } else {
#pragma unroll
            for (int j = 0; j < 4; ++j)
#pragma unroll
                for (int r = 0; r < 8; ++r) s[j][r] = s[j][r] * sc;
        }

        // online softmax: rows live across 16-lane halves -> xor masks 8..1
#pragma unroll
        for (int r = 0; r < 8; ++r) {
            float mx = s[0][r];
#pragma unroll
            for (int j = 1; j < 4; ++j) mx = fmaxf(mx, s[j][r]);
#pragma unroll
            for (int off = 8; off > 0; off >>= 1) mx = fmaxf(mx, __shfl_xor(mx, off, 32));
            const float mnew  = fmaxf(mrun[r], mx);
            const float alpha = __expf(mrun[r] - mnew);
            float rs = 0.0f;
#pragma unroll
            for (int j = 0; j < 4; ++j) {
                const float p = __expf(s[j][r] - mnew);
                s[j][r] = p;
                rs += p;
            }
#pragma unroll
            for (int off = 8; off > 0; off >>= 1) rs += __shfl_xor(rs, off, 32);
            lrun[r] = lrun[r] * alpha + rs;
            mrun[r] = mnew;
#pragma unroll
            for (int n = 0; n < 4; ++n) accO[n][r] *= alpha;
        }

        // stage P (D-layout) -> LDS -> reload in A-fragment layout
#pragma unroll
        for (int j = 0; j < 4; ++j)
#pragma unroll
            for (int r = 0; r < 8; ++r)
                Psm[wv][r + half * 8][j * 16 + lo] = (bf16_t)s[j][r];
        asm volatile("s_wait_dscnt 0" ::: "memory");   // wave-local LDS RAW

#pragma unroll
        for (int kk = 0; kk < 2; ++kk) {
            FragU ap;
            const bf16_t* p = &Psm[wv][lo][kk * 32 + half * 8];
            ap.q[0] = *(const uint4*)(p);
            ap.q[1] = *(const uint4*)(p + 16);
#pragma unroll
            for (int n = 0; n < 4; ++n) {
                FragU bvf;
                const bf16_t* pv = &Vsm[bi][n * 16 + lo][kk * 32 + half * 8];
                bvf.q[0] = *(const uint4*)(pv);
                bvf.q[1] = *(const uint4*)(pv + 16);
                accO[n] = wmma_bf16(ap, bvf, accO[n]);
            }
        }
        __syncthreads();   // everyone done with buf bi before it is refilled
    }

    // O = accO / l   -> attn-out bf16 [B*N][D], column h*64 + n*16 + lo
#pragma unroll
    for (int n = 0; n < 4; ++n) {
#pragma unroll
        for (int r = 0; r < 8; ++r) {
            const int q   = q0 + r + half * 8;
            const int col = h * HDIM + n * 16 + lo;
            const float v = accO[n][r] / lrun[r];
            AOb[((size_t)b * NSEQ + q) * DMODEL + col] = (bf16_t)v;
        }
    }
}

// ---------------------------------------------------------------------------
extern "C" void kernel_launch(void* const* d_in, const int* in_sizes, int n_in,
                              void* d_out, int out_size, void* d_ws, size_t ws_size,
                              hipStream_t stream) {
    const float* x     = (const float*)d_in[0];
    // d_in[1] = causal_mask (unused: mask computed analytically)
    const float* Wqkv  = (const float*)d_in[2];
    const float* bqkv  = (const float*)d_in[3];
    const float* Wproj = (const float*)d_in[4];
    const float* bproj = (const float*)d_in[5];
    float* out = (float*)d_out;

    char* ws = (char*)d_ws;
    size_t off = 0;
    auto alloc = [&](size_t bytes) -> void* {
        void* p = ws + off;
        off += (bytes + 255) & ~(size_t)255;
        return p;
    };
    bf16_t* Xb     = (bf16_t*)alloc((size_t)MROWS * DMODEL * 2);
    bf16_t* WqkvT  = (bf16_t*)alloc((size_t)N3 * DMODEL * 2);
    bf16_t* WprojT = (bf16_t*)alloc((size_t)DMODEL * DMODEL * 2);
    bf16_t* Qb     = (bf16_t*)alloc((size_t)MROWS * DMODEL * 2);
    bf16_t* Kb     = (bf16_t*)alloc((size_t)MROWS * DMODEL * 2);
    bf16_t* Vtb    = (bf16_t*)alloc((size_t)MROWS * DMODEL * 2);
    bf16_t* AOb    = (bf16_t*)alloc((size_t)MROWS * DMODEL * 2);
    (void)ws_size; (void)in_sizes; (void)n_in; (void)out_size;

    {
        int n = MROWS * DMODEL;
        k_f32_to_bf16<<<(n + 255) / 256, 256, 0, stream>>>(x, Xb, n);
    }
    {
        int n = DMODEL * N3;
        k_transpose_bf16<<<(n + 255) / 256, 256, 0, stream>>>(Wqkv, WqkvT, DMODEL, N3);
    }
    {
        int n = DMODEL * DMODEL;
        k_transpose_bf16<<<(n + 255) / 256, 256, 0, stream>>>(Wproj, WprojT, DMODEL, DMODEL);
    }

    k_gemm<0><<<dim3(N3 / 128, MROWS / 128), 256, 0, stream>>>(
        Xb, WqkvT, bqkv, nullptr, Qb, Kb, Vtb, MROWS, N3, DMODEL);

    k_flash<<<dim3(NSEQ / 64, BATCH * NHEAD), 128, 0, stream>>>(Qb, Kb, Vtb, AOb);

    k_gemm<1><<<dim3(DMODEL / 128, MROWS / 128), 256, 0, stream>>>(
        AOb, WprojT, bproj, out, nullptr, nullptr, nullptr, MROWS, DMODEL, DMODEL);
}
